// Spectral_26018911879225
// MI455X (gfx1250) — matrix operand order repaired and checked by reading, not compiled, not run on hardware
//
#include <hip/hip_runtime.h>
#include <hip/hip_bf16.h>
#include <math.h>

typedef __bf16 bf16_t;
typedef __attribute__((ext_vector_type(16))) __bf16 v16bf;
typedef __attribute__((ext_vector_type(8)))  float  v8f;

#if defined(__has_builtin)
#  if __has_builtin(__builtin_amdgcn_global_load_async_to_lds_b128) && \
      __has_builtin(__builtin_amdgcn_s_wait_asynccnt)
#    define USE_ASYNC 1
#  endif
#endif
#ifndef USE_ASYNC
#  define USE_ASYNC 0
#endif

#if USE_ASYNC
typedef int v4i_t __attribute__((vector_size(16)));
typedef __attribute__((address_space(1))) v4i_t* gptr_t;   // global (AS1)
typedef __attribute__((address_space(3))) v4i_t* lptr_t;   // LDS (AS3)
// AS1 pointer value == flat address value for global memory.
__device__ __forceinline__ gptr_t to_gas(const void* p) {
    return (gptr_t)(unsigned long long)p;
}
// Flat LDS address truncates to the LDS byte offset (ISA 10.2 aperture rules);
// AS3 pointers are 32-bit.
__device__ __forceinline__ lptr_t to_lds(void* p) {
    return (lptr_t)(unsigned)(unsigned long long)p;
}
#endif

// ---------------------------------------------------------------- helpers
__device__ __forceinline__ bf16_t f2bf(float f) {
    union { float f; unsigned u; } c; c.f = f;
    unsigned r = c.u + 0x7FFFu + ((c.u >> 16) & 1u);     // round-to-nearest-even
    union { unsigned short s; bf16_t b; } o; o.s = (unsigned short)(r >> 16);
    return o.b;
}

union FragU { uint4 u[2]; v16bf v; };

// ---------------------------------------------------------------- f32 -> bf16
__global__ __launch_bounds__(256)
void cvt_f32_to_bf16(const float* __restrict__ in, bf16_t* __restrict__ out, size_t n) {
    size_t i = ((size_t)blockIdx.x * blockDim.x + threadIdx.x) * 4;
    if (i + 3 < n) {
        float4 v = *(const float4*)(in + i);
        out[i + 0] = f2bf(v.x);
        out[i + 1] = f2bf(v.y);
        out[i + 2] = f2bf(v.z);
        out[i + 3] = f2bf(v.w);
    }
}

// ---------------------------------------------------------------- WMMA GEMM
// C[M,N] (f32) = A[M,K] (bf16 row-major) * W[N,K]^T (bf16 row-major over K)
// Block tile 128 x (32*TN), K-step 32. 8 waves as 4(M) x 2(N); each wave owns
// a 32 x (16*TN) set of 16x16 WMMA tiles. Double-buffered LDS; next tile is
// fetched with gfx1250 async load-to-LDS (ASYNCcnt) while WMMAs run.
#define BM 128
#define BK 32
#define LDA 40   // bf16 elements, +8 pad (80B rows, 16B aligned)
#define LDB 40

template<int TN>
__global__ __launch_bounds__(256)
void gemm_bf16_wmma(const bf16_t* __restrict__ A, const bf16_t* __restrict__ W,
                    float* __restrict__ C, int M, int N, int K) {
    constexpr int BN_  = 32 * TN;
    constexpr int AVEC = (BM  * (BK / 8)) / 256;   // uint4 loads per thread for A tile (=2)
    constexpr int BVEC = (BN_ * (BK / 8)) / 256;   // uint4 loads per thread for B tile (=TN/2)

    __shared__ bf16_t As[2][BM  * LDA];
    __shared__ bf16_t Bs[2][BN_ * LDB];

    const int tid  = threadIdx.x;
    const int lane = tid & 31;
    const int wave = tid >> 5;        // 0..7
    const int wm   = wave & 3;        // 4 waves along M
    const int wn   = wave >> 2;       // 2 waves along N
    const int bm0  = blockIdx.x * BM;
    const int bn0  = blockIdx.y * BN_;
    const int half = lane >> 4;       // 0: lanes 0-15, 1: lanes 16-31
    const int l16  = lane & 15;

    v8f acc[2][TN] = {};

    auto mma_step = [&](int buf) {
        FragU a[2], b[TN];
        // A frag (ISA 16x32 bf16): lanes<16 hold K {0..7,16..23}, lanes>=16 K {8..15,24..31}
        #pragma unroll
        for (int i = 0; i < 2; ++i) {
            int row = wm * 32 + i * 16 + l16;
            int ko  = half ? 8 : 0;
            a[i].u[0] = *(const uint4*)(&As[buf][row * LDA + ko]);
            a[i].u[1] = *(const uint4*)(&As[buf][row * LDA + ko + 16]);
        }
        // B frag (32x16): lane = column n; lanes<16 K {0..7,16..23}, lanes>=16 K {8..15,24..31}
        #pragma unroll
        for (int j = 0; j < TN; ++j) {
            int col = wn * (16 * TN) + j * 16 + l16;
            int ko  = half ? 8 : 0;
            b[j].u[0] = *(const uint4*)(&Bs[buf][col * LDB + ko]);
            b[j].u[1] = *(const uint4*)(&Bs[buf][col * LDB + ko + 16]);
        }
        #pragma unroll
        for (int i = 0; i < 2; ++i)
            #pragma unroll
            for (int j = 0; j < TN; ++j)
                acc[i][j] = __builtin_amdgcn_wmma_f32_16x16x32_bf16(
                    false, a[i].v, false, b[j].v, (short)0, acc[i][j], false, false);
    };

#if USE_ASYNC
    auto issue_async = [&](int buf, int k0) {
        #pragma unroll
        for (int i = 0; i < AVEC; ++i) {
            int idx = tid + i * 256, r = idx >> 2, cv = (idx & 3) * 8;
            __builtin_amdgcn_global_load_async_to_lds_b128(
                to_gas(A + (size_t)(bm0 + r) * K + k0 + cv),
                to_lds(&As[buf][r * LDA + cv]), 0, 0);
        }
        #pragma unroll
        for (int i = 0; i < BVEC; ++i) {
            int idx = tid + i * 256, r = idx >> 2, cv = (idx & 3) * 8;
            __builtin_amdgcn_global_load_async_to_lds_b128(
                to_gas(W + (size_t)(bn0 + r) * K + k0 + cv),
                to_lds(&Bs[buf][r * LDB + cv]), 0, 0);
        }
    };

    issue_async(0, 0);
    __builtin_amdgcn_s_wait_asynccnt(0);
    __syncthreads();
    int cur = 0;
    for (int k0 = 0; k0 < K; k0 += BK) {
        if (k0 + BK < K) issue_async(cur ^ 1, k0 + BK);  // DMA next tile while computing
        mma_step(cur);
        __builtin_amdgcn_s_wait_asynccnt(0);
        __syncthreads();
        cur ^= 1;
    }
#else
    // Fallback: register-staged double buffering (loads batched, no per-element wait)
    uint4 ra[AVEC], rb[BVEC];
    auto load_regs = [&](int k0) {
        #pragma unroll
        for (int i = 0; i < AVEC; ++i) {
            int idx = tid + i * 256, r = idx >> 2, cv = (idx & 3) * 8;
            ra[i] = *(const uint4*)(A + (size_t)(bm0 + r) * K + k0 + cv);
        }
        #pragma unroll
        for (int i = 0; i < BVEC; ++i) {
            int idx = tid + i * 256, r = idx >> 2, cv = (idx & 3) * 8;
            rb[i] = *(const uint4*)(W + (size_t)(bn0 + r) * K + k0 + cv);
        }
    };
    auto store_lds = [&](int buf) {
        #pragma unroll
        for (int i = 0; i < AVEC; ++i) {
            int idx = tid + i * 256, r = idx >> 2, cv = (idx & 3) * 8;
            *(uint4*)(&As[buf][r * LDA + cv]) = ra[i];
        }
        #pragma unroll
        for (int i = 0; i < BVEC; ++i) {
            int idx = tid + i * 256, r = idx >> 2, cv = (idx & 3) * 8;
            *(uint4*)(&Bs[buf][r * LDB + cv]) = rb[i];
        }
    };

    load_regs(0);
    store_lds(0);
    __syncthreads();
    int cur = 0;
    for (int k0 = 0; k0 < K; k0 += BK) {
        bool more = (k0 + BK) < K;
        if (more) load_regs(k0 + BK);     // global loads overlap WMMAs below
        mma_step(cur);
        __syncthreads();
        if (more) store_lds(cur ^ 1);
        __syncthreads();
        cur ^= 1;
    }
#endif

    // C/D layout: VGPR r -> M = r + 8*half, N = l16
    #pragma unroll
    for (int i = 0; i < 2; ++i)
        #pragma unroll
        for (int j = 0; j < TN; ++j) {
            int n = bn0 + wn * (16 * TN) + j * 16 + l16;
            #pragma unroll
            for (int r = 0; r < 8; ++r) {
                int m = bm0 + wm * 32 + i * 16 + r + half * 8;
                C[(size_t)m * N + n] = acc[i][j][r];
            }
        }
}

// ---------------------------------------------------------------- RMS norm (in place, one block per row)
__global__ __launch_bounds__(256)
void rmsnorm_inplace(float* __restrict__ X, int cols) {
    float* p = X + (size_t)blockIdx.x * cols;
    float ss = 0.f;
    for (int i = threadIdx.x; i < cols; i += blockDim.x) { float v = p[i]; ss += v * v; }
    __shared__ float red[8];
    for (int o = 16; o > 0; o >>= 1) ss += __shfl_down(ss, o, 32);
    int lane = threadIdx.x & 31, wv = threadIdx.x >> 5;
    if (lane == 0) red[wv] = ss;
    __syncthreads();
    if (wv == 0) {
        float v = (lane < 8) ? red[lane] : 0.f;
        for (int o = 4; o > 0; o >>= 1) v += __shfl_down(v, o, 32);
        if (lane == 0) red[0] = v;
    }
    __syncthreads();
    float scale = rsqrtf(red[0] / (float)cols + 1.1920929e-07f);
    for (int i = threadIdx.x; i < cols; i += blockDim.x) p[i] *= scale;
}

// ---------------------------------------------------------------- fused 12-level butterfly in LDS
#define CD 4
__global__ __launch_bounds__(512)
void butterfly_fused(const float* __restrict__ theta, const float* __restrict__ phi,
                     bf16_t* __restrict__ out, int Tn, int H, int Dh, int L) {
    extern __shared__ float smem[];
    float* buf0 = smem;                        // Tn*CD
    float* buf1 = smem + Tn * CD;              // Tn*CD
    float* csC  = smem + 2 * Tn * CD;          // Tn
    float* csS  = csC + Tn;                    // Tn

    const int b   = blockIdx.z;
    const int h   = blockIdx.y;
    const int dh0 = blockIdx.x * CD;
    const int D   = H * Dh;
    const int tid = threadIdx.x;
    const float inv_sqrt2 = 0.70710678118654752440f;

    for (int idx = tid; idx < Tn * CD; idx += blockDim.x) {
        int t = idx / CD, j = idx % CD;
        buf0[idx] = theta[(size_t)(b * Tn + t) * D + h * Dh + dh0 + j];
    }

    float* cur = buf0; float* nxt = buf1;
    for (int l = 0; l < L; ++l) {
        __syncthreads();
        for (int t = tid; t < Tn; t += blockDim.x) {
            float ph = phi[(size_t)(b * Tn + t) * (L * H) + l * H + h];
            csC[t] = __cosf(ph) * inv_sqrt2;
            csS[t] = __sinf(ph) * inv_sqrt2;
        }
        __syncthreads();
        const int step = 1 << l;
        for (int idx = tid; idx < Tn * CD; idx += blockDim.x) {
            int t = idx / CD;
            float prev = (t >= step) ? cur[idx - step * CD] : 0.f;
            nxt[idx] = csC[t] * cur[idx] + csS[t] * prev;
        }
        float* tmp = cur; cur = nxt; nxt = tmp;
    }
    __syncthreads();
    for (int idx = tid; idx < Tn * CD; idx += blockDim.x) {
        int t = idx / CD, j = idx % CD;
        out[(size_t)(b * Tn + t) * D + h * Dh + dh0 + j] = f2bf(cur[idx]);
    }
}

// ---------------------------------------------------------------- launch
extern "C" void kernel_launch(void* const* d_in, const int* in_sizes, int n_in,
                              void* d_out, int out_size, void* d_ws, size_t ws_size,
                              hipStream_t stream) {
    (void)in_sizes; (void)n_in; (void)out_size; (void)ws_size;
    const int B = 2, T = 4096, Dm = 2048, H = 16, L = 12;
    const int M = B * T;               // 8192
    const int E = L * H;               // 192

    const float* x     = (const float*)d_in[0];
    const float* W_tok = (const float*)d_in[1];
    const float* W_lvl = (const float*)d_in[2];
    const float* W_out = (const float*)d_in[3];
    float* y = (float*)d_out;

    char* ws = (char*)d_ws;
    bf16_t* xb      = (bf16_t*)(ws);                         size_t o = (size_t)M * Dm * 2;
    bf16_t* wtok_b  = (bf16_t*)(ws + o);                     o += (size_t)Dm * Dm * 2;
    bf16_t* wlvl_b  = (bf16_t*)(ws + o);                     o += (size_t)E * Dm * 2;
    bf16_t* wout_b  = (bf16_t*)(ws + o);                     o += (size_t)Dm * Dm * 2;
    float*  theta   = (float*)(ws + o);                      o += (size_t)M * Dm * 4;
    float*  phi     = (float*)(ws + o);                      o += (size_t)M * E * 4;
    bf16_t* thetabf = (bf16_t*)(ws + o);

    // 1) convert inputs to bf16
    {
        size_t n;
        n = (size_t)M * Dm;  cvt_f32_to_bf16<<<(unsigned)(n / 1024), 256, 0, stream>>>(x, xb, n);
        n = (size_t)Dm * Dm; cvt_f32_to_bf16<<<(unsigned)(n / 1024), 256, 0, stream>>>(W_tok, wtok_b, n);
        n = (size_t)E * Dm;  cvt_f32_to_bf16<<<(unsigned)(n / 1024), 256, 0, stream>>>(W_lvl, wlvl_b, n);
        n = (size_t)Dm * Dm; cvt_f32_to_bf16<<<(unsigned)(n / 1024), 256, 0, stream>>>(W_out, wout_b, n);
    }

    // 2) theta_raw = x @ W_tok^T ; phi_raw = x @ W_lvl^T
    gemm_bf16_wmma<4><<<dim3(M / BM, Dm / 128), 256, 0, stream>>>(xb, wtok_b, theta, M, Dm, Dm);
    gemm_bf16_wmma<2><<<dim3(M / BM, E / 64),   256, 0, stream>>>(xb, wlvl_b, phi,   M, E,  Dm);

    // 3) RMS norms (in place)
    rmsnorm_inplace<<<M, 256, 0, stream>>>(theta, Dm);
    rmsnorm_inplace<<<M, 256, 0, stream>>>(phi, E);

    // 4) fused 12-level butterfly; emits bf16 activations for the final GEMM
    {
        size_t shmem = (size_t)(2 * T * CD + 2 * T) * sizeof(float);   // 160 KB
        butterfly_fused<<<dim3((Dm / H) / CD, H, B), 512, shmem, stream>>>(
            theta, phi, thetabf, T, H, Dm / H, L);
    }

    // 5) y = theta_final @ W_out^T  (f32 out)
    gemm_bf16_wmma<4><<<dim3(M / BM, Dm / 128), 256, 0, stream>>>(thetabf, wout_b, y, M, Dm, Dm);
}
